// NeighborhoodAttention2D_1726576854815
// MI455X (gfx1250) — matrix operand order repaired and checked by reading, not compiled, not run on hardware
//
#include <hip/hip_runtime.h>

typedef unsigned short u16;
typedef unsigned int   u32;

typedef __attribute__((ext_vector_type(16))) __bf16 v16bf;
typedef __attribute__((ext_vector_type(8)))  float  v8f;
typedef __attribute__((ext_vector_type(8)))  u32    v8u;

#define BDIM   8
#define HTOT   56
#define WTOT   56
#define HWP    3136          // 56*56
#define CIN    512
#define NH     16
#define HD     32
#define KS     7
#define RPW    13            // 2*KS-1
#define NPIX   25088         // BDIM*HWP
#define QSCALE 0.17677669529663687f   // HD^-0.5

#define LDA 48               // LDS row stride (bf16), 96B rows -> 16B-aligned chunks
#define LDB 48
#define KSTEP 32
#define NSTEPS (CIN / KSTEP) // 16

// ---------- async global->LDS (gfx1250 ASYNCcnt path), guarded ----------
#if defined(__has_builtin)
#if __has_builtin(__builtin_amdgcn_global_load_async_to_lds_b128) && \
    __has_builtin(__builtin_amdgcn_s_wait_asynccnt)
#define HAVE_ASYNC_LDS 1
#endif
#endif

#ifdef HAVE_ASYNC_LDS
// builtin expects: (v4i addrspace(1)*, v4i addrspace(3)*, imm offset, imm cpol)
typedef int gcc_v4i __attribute__((vector_size(4 * sizeof(int))));
typedef __attribute__((address_space(1))) gcc_v4i* g_v4i_p;
typedef __attribute__((address_space(3))) gcc_v4i* l_v4i_p;
__device__ __forceinline__ void async_ld_b128(const void* g, void* l) {
  // flat shared address: low 32 bits are the LDS byte offset (ISA aperture rule)
  __builtin_amdgcn_global_load_async_to_lds_b128(
      (g_v4i_p)(unsigned long long)(size_t)g,
      (l_v4i_p)(unsigned)(size_t)l, 0, 0);
}
__device__ __forceinline__ void async_wait0() { __builtin_amdgcn_s_wait_asynccnt(0); }
#endif

// ---------- bf16 helpers ----------
__device__ __forceinline__ u16 f2bf(float f) {
  union { float f; u32 u; } x; x.f = f;
  u32 r = x.u + 0x7FFFu + ((x.u >> 16) & 1u);   // round-to-nearest-even
  return (u16)(r >> 16);
}
__device__ __forceinline__ float bflo(u32 u) {
  union { u32 v; float f; } x; x.v = u << 16; return x.f;
}
__device__ __forceinline__ float bfhi(u32 u) {
  union { u32 v; float f; } x; x.v = u & 0xFFFF0000u; return x.f;
}

// ---------- WMMA fragment loaders (CDNA5 wave32 layouts) ----------
// A 16x32 bf16: lane%16 = M row; K pairs interleaved: K(2j) = 2j + (lane/16)*8 + (j>=4 ? 8 : 0)
__device__ __forceinline__ v16bf load_a_frag(const u16* As, int ml, int lh) {
  v8u au;
  #pragma unroll
  for (int jj = 0; jj < 8; ++jj) {
    int k = 2 * jj + lh * 8 + ((jj >= 4) ? 8 : 0);
    au[jj] = *(const u32*)&As[ml * LDA + k];
  }
  return __builtin_bit_cast(v16bf, au);
}
// B 32x16 bf16: lane%16 = N column; K = 8*lh + 0..7, then 16 + 8*lh + 0..7 (same as A)
__device__ __forceinline__ v16bf load_b_frag(const u16* Bs, int nl, int lh) {
  uint4 q0 = *(const uint4*)&Bs[nl * LDB + lh * 8];
  uint4 q1 = *(const uint4*)&Bs[nl * LDB + lh * 8 + 16];
  v8u bu;
  bu[0]=q0.x; bu[1]=q0.y; bu[2]=q0.z; bu[3]=q0.w;
  bu[4]=q1.x; bu[5]=q1.y; bu[6]=q1.z; bu[7]=q1.w;
  return __builtin_bit_cast(v16bf, bu);
}

// ---------- kernel 1: weight fp32 -> bf16 ----------
__global__ void cvt_weights(const float* __restrict__ qkv_w, const float* __restrict__ proj_w,
                            u16* __restrict__ wq, u16* __restrict__ wp) {
  int i = blockIdx.x * 256 + threadIdx.x;        // grid covers 1536*512 = 786432
  if (i < 3 * CIN * CIN) wq[i] = f2bf(qkv_w[i]);
  if (i < CIN * CIN)     wp[i] = f2bf(proj_w[i]);
}

// ---------- kernel 2: QKV GEMM (M=pixels, N=3C, K=C), bf16 WMMA, f32 accum ----------
// block tile 64(M) x 256(N); 8 waves = 2(M) x 4(N); wave tile 32x64 = 2x4 WMMA frags.
// Double-buffered LDS; B tile filled via async global->LDS when available.
__launch_bounds__(256)
__global__ void qkv_gemm(const float* __restrict__ x, const u16* __restrict__ wq,
                         const float* __restrict__ qkv_b,
                         u16* __restrict__ qb, u16* __restrict__ kb, u16* __restrict__ vb) {
  __shared__ u16 As[2][64 * LDA];     // 12 KB
  __shared__ u16 Bs[2][256 * LDB];    // 48 KB

  const int tid   = threadIdx.x;
  const int lane  = tid & 31;
  const int wave  = tid >> 5;
  const int lr    = lane & 15;
  const int lh    = lane >> 4;
  const int waveM = wave & 1;
  const int waveN = wave >> 1;

  const int m0   = blockIdx.y * 64;      // pixel base (3136 % 64 == 0 -> one image per tile)
  const int n0   = blockIdx.x * 256;     // qkv channel base
  const int img  = m0 / HWP;
  const int pix0 = m0 - img * HWP;

  const int a_k  = tid >> 3;             // 0..31
  const int a_pg = (tid & 7) * 8;        // 8-pixel group
  // B loader: thread t fills full row t (256 rows x 32 K = 4 x b128 per thread)

  v8f acc[2][4] = {};
  int s = 0;

  // ---- prologue: fill stage 0 ----
  {
    const float* xp = x + ((size_t)(img * CIN + a_k)) * HWP + pix0 + a_pg;
    float4 f0 = *(const float4*)xp;
    float4 f1 = *(const float4*)(xp + 4);
    u16* ad = &As[0][a_pg * LDA + a_k];
    ad[0*LDA]=f2bf(f0.x); ad[1*LDA]=f2bf(f0.y); ad[2*LDA]=f2bf(f0.z); ad[3*LDA]=f2bf(f0.w);
    ad[4*LDA]=f2bf(f1.x); ad[5*LDA]=f2bf(f1.y); ad[6*LDA]=f2bf(f1.z); ad[7*LDA]=f2bf(f1.w);
    const u16* wr = wq + (size_t)(n0 + tid) * CIN;
#ifdef HAVE_ASYNC_LDS
    #pragma unroll
    for (int c = 0; c < 32; c += 8) async_ld_b128(wr + c, &Bs[0][tid * LDB + c]);
    async_wait0();
#else
    #pragma unroll
    for (int c = 0; c < 32; c += 8)
      *(uint4*)&Bs[0][tid * LDB + c] = *(const uint4*)(wr + c);
#endif
  }

  for (int step = 0; step < NSTEPS; ++step) {
    __syncthreads();
    const int kk  = step * KSTEP;
    const bool more = (step + 1 < NSTEPS);

    // issue next-stage fills early
    float4 f0, f1;
    if (more) {
      const float* xp = x + ((size_t)(img * CIN + kk + KSTEP + a_k)) * HWP + pix0 + a_pg;
      f0 = *(const float4*)xp;
      f1 = *(const float4*)(xp + 4);
      if (step + 2 < NSTEPS)
        __builtin_prefetch(xp + (size_t)KSTEP * HWP, 0, 0);   // global_prefetch_b8
      const u16* wr = wq + (size_t)(n0 + tid) * CIN + kk + KSTEP;
#ifdef HAVE_ASYNC_LDS
      #pragma unroll
      for (int c = 0; c < 32; c += 8) async_ld_b128(wr + c, &Bs[s ^ 1][tid * LDB + c]);
#else
      #pragma unroll
      for (int c = 0; c < 32; c += 8)
        *(uint4*)&Bs[s ^ 1][tid * LDB + c] = *(const uint4*)(wr + c);
#endif
    }

    // ---- compute stage s ----
    v16bf af[2], bf[4];
    #pragma unroll
    for (int mi = 0; mi < 2; ++mi)
      af[mi] = load_a_frag(As[s], waveM * 32 + mi * 16 + lr, lh);
    #pragma unroll
    for (int ni = 0; ni < 4; ++ni)
      bf[ni] = load_b_frag(Bs[s], waveN * 64 + ni * 16 + lr, lh);
    #pragma unroll
    for (int mi = 0; mi < 2; ++mi)
      #pragma unroll
      for (int ni = 0; ni < 4; ++ni)
        acc[mi][ni] = __builtin_amdgcn_wmma_f32_16x16x32_bf16(
            false, af[mi], false, bf[ni], (short)0, acc[mi][ni], false, false);

    // ---- store next-stage A (f32->bf16 transpose), finish async B ----
    if (more) {
      u16* ad = &As[s ^ 1][a_pg * LDA + a_k];
      ad[0*LDA]=f2bf(f0.x); ad[1*LDA]=f2bf(f0.y); ad[2*LDA]=f2bf(f0.z); ad[3*LDA]=f2bf(f0.w);
      ad[4*LDA]=f2bf(f1.x); ad[5*LDA]=f2bf(f1.y); ad[6*LDA]=f2bf(f1.z); ad[7*LDA]=f2bf(f1.w);
#ifdef HAVE_ASYNC_LDS
      async_wait0();
#endif
    }
    s ^= 1;
  }

  // ---- epilogue: bias, q-scale, scatter to (b, head, pix, hd) bf16 ----
  #pragma unroll
  for (int mi = 0; mi < 2; ++mi) {
    #pragma unroll
    for (int ni = 0; ni < 4; ++ni) {
      const int n    = n0 + waveN * 64 + ni * 16 + lr;   // 0..1535
      const float bs = qkv_b[n];
      const int sel  = n >> 9;                            // 0=q 1=k 2=v
      const int ch   = n & 511;
      u16* dst = (sel == 0) ? qb : ((sel == 1) ? kb : vb);
      const float scl = (sel == 0) ? QSCALE : 1.0f;
      const int head = ch >> 5, hd = ch & 31;
      #pragma unroll
      for (int r = 0; r < 8; ++r) {
        const int m   = m0 + waveM * 32 + mi * 16 + lh * 8 + r;
        const int pix = m - img * HWP;
        float val = (acc[mi][ni][r] + bs) * scl;
        dst[((size_t)(img * NH + head) * HWP + pix) * HD + hd] = f2bf(val);
      }
    }
  }
}

// ---------- kernel 3: neighborhood attention, thread = (b, head, pixel) ----------
__launch_bounds__(256)
__global__ void natten2d(const u16* __restrict__ qb, const u16* __restrict__ kb,
                         const u16* __restrict__ vb, const float* __restrict__ rpb,
                         u16* __restrict__ ao) {
  const int idx  = blockIdx.x * 256 + threadIdx.x;  // 128*3136 = 401408 exact
  const int bh   = idx / HWP;                       // b*NH + head
  const int pix  = idx - bh * HWP;
  const int head = bh & (NH - 1);
  const int i    = pix / WTOT;
  const int j    = pix - i * WTOT;
  const size_t base = (size_t)bh * HWP * HD;

  float q[HD];
  {
    const u16* qp = qb + base + (size_t)pix * HD;
    #pragma unroll
    for (int c4 = 0; c4 < 4; ++c4) {
      uint4 u = *(const uint4*)(qp + c4 * 8);
      q[c4*8+0]=bflo(u.x); q[c4*8+1]=bfhi(u.x); q[c4*8+2]=bflo(u.y); q[c4*8+3]=bfhi(u.y);
      q[c4*8+4]=bflo(u.z); q[c4*8+5]=bfhi(u.z); q[c4*8+6]=bflo(u.w); q[c4*8+7]=bfhi(u.w);
    }
  }

  int si = i - (KS/2); si = si < 0 ? 0 : (si > HTOT-KS ? HTOT-KS : si);
  int sj = j - (KS/2); sj = sj < 0 ? 0 : (sj > WTOT-KS ? WTOT-KS : sj);
  const int bi0 = si - i + (KS - 1);
  const int bj0 = sj - j + (KS - 1);
  const float* rp = rpb + head * RPW * RPW;

  float logits[KS * KS];
  #pragma unroll
  for (int ka = 0; ka < KS; ++ka) {
    #pragma unroll
    for (int kq = 0; kq < KS; ++kq) {
      const u16* kp = kb + base + (size_t)((si + ka) * WTOT + (sj + kq)) * HD;
      float a = 0.f;
      #pragma unroll
      for (int c4 = 0; c4 < 4; ++c4) {
        uint4 u = *(const uint4*)(kp + c4 * 8);
        a += q[c4*8+0]*bflo(u.x) + q[c4*8+1]*bfhi(u.x)
           + q[c4*8+2]*bflo(u.y) + q[c4*8+3]*bfhi(u.y)
           + q[c4*8+4]*bflo(u.z) + q[c4*8+5]*bfhi(u.z)
           + q[c4*8+6]*bflo(u.w) + q[c4*8+7]*bfhi(u.w);
      }
      logits[ka * KS + kq] = a + rp[(bi0 + ka) * RPW + (bj0 + kq)];
    }
  }

  float mx = logits[0];
  #pragma unroll
  for (int t = 1; t < KS * KS; ++t) mx = fmaxf(mx, logits[t]);
  float ssum = 0.f;
  #pragma unroll
  for (int t = 0; t < KS * KS; ++t) { float e = __expf(logits[t] - mx); logits[t] = e; ssum += e; }
  const float inv = 1.f / ssum;

  float out[HD] = {};
  #pragma unroll
  for (int ka = 0; ka < KS; ++ka) {
    #pragma unroll
    for (int kq = 0; kq < KS; ++kq) {
      const float w = logits[ka * KS + kq];
      const u16* vp = vb + base + (size_t)((si + ka) * WTOT + (sj + kq)) * HD;
      #pragma unroll
      for (int c4 = 0; c4 < 4; ++c4) {
        uint4 u = *(const uint4*)(vp + c4 * 8);
        out[c4*8+0] += w*bflo(u.x); out[c4*8+1] += w*bfhi(u.x);
        out[c4*8+2] += w*bflo(u.y); out[c4*8+3] += w*bfhi(u.y);
        out[c4*8+4] += w*bflo(u.z); out[c4*8+5] += w*bfhi(u.z);
        out[c4*8+6] += w*bflo(u.w); out[c4*8+7] += w*bfhi(u.w);
      }
    }
  }

  u16* op = ao + (size_t)((bh >> 4) * HWP + pix) * CIN + head * HD;
  #pragma unroll
  for (int t = 0; t < HD; ++t) op[t] = f2bf(out[t] * inv);
}

// ---------- kernel 4: projection GEMM, TRANSPOSED: M=channels(512), N=pixels(25088) ----------
// y.T[c, pixel] = sum_k proj_w[c,k] * attn_out[pixel,k] -> pixel-major stores into (B,C,H,W)
__launch_bounds__(256)
__global__ void proj_gemm(const u16* __restrict__ wp, const u16* __restrict__ ao,
                          const float* __restrict__ proj_b, float* __restrict__ out) {
  __shared__ u16 As[2][64 * LDA];
  __shared__ u16 Bs[2][256 * LDB];

  const int tid   = threadIdx.x;
  const int lane  = tid & 31;
  const int wave  = tid >> 5;
  const int lr    = lane & 15;
  const int lh    = lane >> 4;
  const int waveM = wave & 1;
  const int waveN = wave >> 1;

  const int m0 = blockIdx.y * 64;       // channel base
  const int n0 = blockIdx.x * 256;      // pixel base

  const int a_rw = tid >> 2;            // 0..63
  const int a_c  = (tid & 3) * 8;       // K chunk

  v8f acc[2][4] = {};
  int s = 0;

  // ---- prologue: fill stage 0 (both operands bf16 -> async-capable) ----
  {
    const u16* ar = wp + (size_t)(m0 + a_rw) * CIN + a_c;
    const u16* br = ao + (size_t)(n0 + tid) * CIN;
#ifdef HAVE_ASYNC_LDS
    async_ld_b128(ar, &As[0][a_rw * LDA + a_c]);
    #pragma unroll
    for (int c = 0; c < 32; c += 8) async_ld_b128(br + c, &Bs[0][tid * LDB + c]);
    async_wait0();
#else
    *(uint4*)&As[0][a_rw * LDA + a_c] = *(const uint4*)ar;
    #pragma unroll
    for (int c = 0; c < 32; c += 8)
      *(uint4*)&Bs[0][tid * LDB + c] = *(const uint4*)(br + c);
#endif
  }

  for (int step = 0; step < NSTEPS; ++step) {
    __syncthreads();
    const int kk  = step * KSTEP;
    const bool more = (step + 1 < NSTEPS);

    if (more) {
      const u16* ar = wp + (size_t)(m0 + a_rw) * CIN + kk + KSTEP + a_c;
      const u16* br = ao + (size_t)(n0 + tid) * CIN + kk + KSTEP;
#ifdef HAVE_ASYNC_LDS
      async_ld_b128(ar, &As[s ^ 1][a_rw * LDA + a_c]);
      #pragma unroll
      for (int c = 0; c < 32; c += 8) async_ld_b128(br + c, &Bs[s ^ 1][tid * LDB + c]);
#else
      *(uint4*)&As[s ^ 1][a_rw * LDA + a_c] = *(const uint4*)ar;
      #pragma unroll
      for (int c = 0; c < 32; c += 8)
        *(uint4*)&Bs[s ^ 1][tid * LDB + c] = *(const uint4*)(br + c);
#endif
    }

    v16bf af[2], bf[4];
    #pragma unroll
    for (int mi = 0; mi < 2; ++mi)
      af[mi] = load_a_frag(As[s], waveM * 32 + mi * 16 + lr, lh);
    #pragma unroll
    for (int ni = 0; ni < 4; ++ni)
      bf[ni] = load_b_frag(Bs[s], waveN * 64 + ni * 16 + lr, lh);
    #pragma unroll
    for (int mi = 0; mi < 2; ++mi)
      #pragma unroll
      for (int ni = 0; ni < 4; ++ni)
        acc[mi][ni] = __builtin_amdgcn_wmma_f32_16x16x32_bf16(
            false, af[mi], false, bf[ni], (short)0, acc[mi][ni], false, false);

#ifdef HAVE_ASYNC_LDS
    if (more) async_wait0();
#endif
    s ^= 1;
  }

  #pragma unroll
  for (int mi = 0; mi < 2; ++mi) {
    #pragma unroll
    for (int ni = 0; ni < 4; ++ni) {
      const int pixel = n0 + waveN * 64 + ni * 16 + lr;   // N = pixel
      const int bimg  = pixel / HWP;
      const int pix   = pixel - bimg * HWP;
      #pragma unroll
      for (int r = 0; r < 8; ++r) {
        const int c = m0 + waveM * 32 + mi * 16 + lh * 8 + r;  // M = channel
        out[((size_t)(bimg * CIN + c)) * HWP + pix] = acc[mi][ni][r] + proj_b[c];
      }
    }
  }
}

extern "C" void kernel_launch(void* const* d_in, const int* in_sizes, int n_in,
                              void* d_out, int out_size, void* d_ws, size_t ws_size,
                              hipStream_t stream) {
  const float* x      = (const float*)d_in[0];
  const float* qkv_w  = (const float*)d_in[1];
  const float* qkv_b  = (const float*)d_in[2];
  const float* rpb    = (const float*)d_in[3];
  const float* proj_w = (const float*)d_in[4];
  const float* proj_b = (const float*)d_in[5];
  float* out = (float*)d_out;

  char* ws = (char*)d_ws;
  u16* wq  = (u16*)(ws);                               // 1536*512*2  = 0x180000
  u16* wpj = (u16*)(ws + 0x180000);                    //  512*512*2  = 0x080000
  u16* qb  = (u16*)(ws + 0x200000);                    // 25088*512*2 = 0x1880000
  u16* kbuf= (u16*)(ws + 0x200000 + 0x1880000ULL);
  u16* vbuf= (u16*)(ws + 0x200000 + 2*0x1880000ULL);
  u16* ao  = (u16*)(ws + 0x200000 + 3*0x1880000ULL);   // total 100 MB

  cvt_weights<<<3072, 256, 0, stream>>>(qkv_w, proj_w, wq, wpj);
  qkv_gemm<<<dim3(6, 392), 256, 0, stream>>>(x, wq, qkv_b, qb, kbuf, vbuf);
  natten2d<<<1568, 256, 0, stream>>>(qb, kbuf, vbuf, rpb, ao);
  proj_gemm<<<dim3(98, 8), 256, 0, stream>>>(wpj, ao, proj_b, out);
}
